// RNNPitchPredictorModel_23132693856840
// MI455X (gfx1250) — matrix operand order repaired and checked by reading, not compiled, not run on hardware
//
#include <hip/hip_runtime.h>
#include <hip/hip_bf16.h>
#include <math.h>

// ---------------------------------------------------------------------------
// LSTM (B=32, T=512, D=512, H=1024, O=128) for MI455X / gfx1250, wave32.
//
//   1) x -> bf16; Wi/Wh/Wd -> bf16 transposed ([N][K], K-contiguous) so all
//      WMMA fragments fill with contiguous per-lane 16B/32B vector loads.
//   2) xz = x@Wi + bh : dense v_wmma_f32_16x16x32_bf16 GEMM with a ping-pong
//      (unroll-2) fragment pipeline: no register-rotation moves, loads for
//      k+32 retire under the WMMAs for k (partial s_wait_loadcnt).
//      Output in [T][B][4H] f32 so each timestep slice is contiguous.
//   3) Persistent scan: 128 blocks x 8 waves. Block (m,nj) owns batch rows
//      m*16..+15 and h-cols nj*16..+15 for ALL FOUR gates. K=1024 split 8
//      ways across waves; Wh B-fragments (128 VGPRs/wave) preloaded once
//      for all 512 steps (no Wh memory traffic in steady state). Cell state
//      c lives in wave-0 registers for the whole scan. xz for step t+1 is
//      loaded into registers during step t and completes under the grid
//      barrier. Activations use hardware v_tanh_f32.
//   4) One-block WMMA kernel for logits (same ping-pong pipeline).
// ---------------------------------------------------------------------------

typedef __attribute__((ext_vector_type(16))) __bf16 v16bf;
typedef __attribute__((ext_vector_type(8)))  __bf16 v8bf;
typedef __attribute__((ext_vector_type(8)))  float  v8f;

#define NSCAN_BLOCKS 128
#define T_STEPS 512
#define HDIM 1024
#define DDIM 512
#define BDIM 32
#define G4H  4096
#define ODIM 128

// ---- workspace layout (bytes), all 256-aligned ----------------------------
#define OFF_XB   ((size_t)0)                 // x bf16                 16 MB
#define OFF_WIT  ((size_t)16777216)          // WiT bf16                4 MB
#define OFF_WHT  ((size_t)20971520)          // WhT bf16                8 MB
#define OFF_WDT  ((size_t)29360128)          // WdT bf16              256 KB
#define OFF_H    ((size_t)29622272)          // h bf16                 64 KB
#define OFF_CTR  ((size_t)29818880)          // barrier counter
#define OFF_XZ   ((size_t)29819136)          // xz f32 [T][B][4H]     256 MB

// ---- fragment loaders (match CDNA5 ISA 16-bit layouts) ---------
__device__ __forceinline__ v16bf load_a16x32(const __bf16* Atile, int lda, int lane) {
    int m = lane & 15, half = lane >> 4;
    const __bf16* p = Atile + (size_t)m * lda + half * 8;
    v8bf lo = *(const v8bf*)(p);
    v8bf hi = *(const v8bf*)(p + 16);
    return __builtin_shufflevector(lo, hi, 0,1,2,3,4,5,6,7,8,9,10,11,12,13,14,15);
}

__device__ __forceinline__ v16bf load_b32x16T(const __bf16* BTtile, int ldb, int lane) {
    int n = lane & 15, half = lane >> 4;
    const __bf16* p = BTtile + (size_t)n * ldb + half * 8;
    v8bf lo = *(const v8bf*)(p);
    v8bf hi = *(const v8bf*)(p + 16);
    return __builtin_shufflevector(lo, hi, 0,1,2,3,4,5,6,7,8,9,10,11,12,13,14,15);
}

// hardware v_tanh_f32 on gfx1250 (TRANS op); fallback for older toolchains
#if defined(__has_builtin)
#if __has_builtin(__builtin_amdgcn_tanh_f32)
#define FAST_TANH(x) __builtin_amdgcn_tanh_f32(x)
#endif
#endif
#ifndef FAST_TANH
#define FAST_TANH(x) tanhf(x)
#endif

__device__ __forceinline__ float fast_sigmoid(float x) {
    // sigmoid(x) = 0.5 + 0.5*tanh(x/2): 1 TRANS + FMA
    return fmaf(0.5f, FAST_TANH(0.5f * x), 0.5f);
}

#define WMMA_BF16(A, B, C) __builtin_amdgcn_wmma_f32_16x16x32_bf16( \
        false, (A), false, (B), (short)0, (C), false, false)

// ---------------------------------------------------------------------------
__global__ void k_init_state(__bf16* h, unsigned* ctr) {
    int idx = blockIdx.x * blockDim.x + threadIdx.x;
    if (idx < BDIM * HDIM) h[idx] = (__bf16)0.0f;
    if (idx == 0) *ctr = 0u;
}

__global__ void k_cvt_x(const float* __restrict__ x, __bf16* __restrict__ xb, int n) {
    int idx = blockIdx.x * blockDim.x + threadIdx.x;
    if (idx < n) xb[idx] = (__bf16)x[idx];
}

__global__ void k_transpose_cvt(const float* __restrict__ W, __bf16* __restrict__ WT,
                                int rows, int cols) {
    int idx = blockIdx.x * blockDim.x + threadIdx.x;
    int total = rows * cols;
    if (idx < total) {
        int n = idx / rows;
        int k = idx - n * rows;
        WT[(size_t)n * rows + k] = (__bf16)W[(size_t)k * cols + n];
    }
}

// ---------------------------------------------------------------------------
// xz[t][b][4H] = x @ Wi + bh. Ping-pong (unroll-2) pipeline: even/odd
// k-steps use disjoint register sets -> zero rotation moves.
__global__ void __launch_bounds__(256)
k_gemm_xz(const __bf16* __restrict__ xb, const __bf16* __restrict__ WiT,
          const float* __restrict__ bh, float* __restrict__ xz) {
    int lane = threadIdx.x & 31;
    int wid  = blockIdx.x * 8 + (threadIdx.x >> 5);
    int m_tile  = wid >> 6;
    int n_strip = wid & 63;

    const __bf16* Abase = xb + (size_t)(m_tile * 16) * DDIM;
    const __bf16* Bbase = WiT + (size_t)(n_strip * 64) * DDIM;

    v8f acc[4] = {};
    v16bf a0, a1, b0[4], b1[4];

    a0 = load_a16x32(Abase, DDIM, lane);
#pragma unroll
    for (int j = 0; j < 4; ++j)
        b0[j] = load_b32x16T(Bbase + (size_t)(j * 16) * DDIM, DDIM, lane);

    for (int k0 = 0; k0 < DDIM - 64; k0 += 64) {
        // even step: prefetch k0+32 into buf1, consume buf0 (k0)
        a1 = load_a16x32(Abase + k0 + 32, DDIM, lane);
#pragma unroll
        for (int j = 0; j < 4; ++j)
            b1[j] = load_b32x16T(Bbase + (size_t)(j * 16) * DDIM + k0 + 32, DDIM, lane);
#pragma unroll
        for (int j = 0; j < 4; ++j)
            acc[j] = WMMA_BF16(a0, b0[j], acc[j]);

        // odd step: prefetch k0+64 into buf0, consume buf1 (k0+32)
        a0 = load_a16x32(Abase + k0 + 64, DDIM, lane);
#pragma unroll
        for (int j = 0; j < 4; ++j)
            b0[j] = load_b32x16T(Bbase + (size_t)(j * 16) * DDIM + k0 + 64, DDIM, lane);
#pragma unroll
        for (int j = 0; j < 4; ++j)
            acc[j] = WMMA_BF16(a1, b1[j], acc[j]);
    }
    // tail: k0 = DDIM-64 -> consume buf0, prefetch+consume DDIM-32
    a1 = load_a16x32(Abase + DDIM - 32, DDIM, lane);
#pragma unroll
    for (int j = 0; j < 4; ++j)
        b1[j] = load_b32x16T(Bbase + (size_t)(j * 16) * DDIM + DDIM - 32, DDIM, lane);
#pragma unroll
    for (int j = 0; j < 4; ++j)
        acc[j] = WMMA_BF16(a0, b0[j], acc[j]);
#pragma unroll
    for (int j = 0; j < 4; ++j)
        acc[j] = WMMA_BF16(a1, b1[j], acc[j]);

    int nl = lane & 15, half = lane >> 4;
#pragma unroll
    for (int j = 0; j < 4; ++j) {
        int col  = n_strip * 64 + j * 16 + nl;
        float bias = bh[col];
#pragma unroll
        for (int i = 0; i < 8; ++i) {
            int r = m_tile * 16 + half * 8 + i;
            int b = r >> 9;
            int t = r & 511;
            xz[(size_t)(t * BDIM + b) * G4H + col] = acc[j][i] + bias;
        }
    }
}

// ---------------------------------------------------------------------------
// Persistent scan. 128 blocks x 256 threads (8 waves), 1 block per WGP.
__global__ void __launch_bounds__(256, 1)
k_lstm_scan(const float* __restrict__ xz, const __bf16* __restrict__ WhT,
            __bf16* __restrict__ h, unsigned* __restrict__ ctr) {
    __shared__ float red[8][4][8][32];   // [k-wave][gate][row-i][lane]  32 KB

    int lane = threadIdx.x & 31;
    int wv   = threadIdx.x >> 5;
    int m    = blockIdx.x & 1;
    int nj   = blockIdx.x >> 1;
    int nl   = lane & 15, half = lane >> 4;
    int kbase = wv * 128;
    int col  = nj * 16 + nl;

    // Wh B-fragments: 4 gates x 4 k-steps = 128 VGPRs, resident for all steps
    v16bf bfr[4][4];
#pragma unroll
    for (int g = 0; g < 4; ++g)
#pragma unroll
        for (int ks = 0; ks < 4; ++ks)
            bfr[g][ks] = load_b32x16T(
                WhT + (size_t)(g * HDIM + nj * 16) * HDIM + kbase + ks * 32,
                HDIM, lane);

    // cell state lives in wave-0 registers for the entire scan
    float creg[8] = {0.f, 0.f, 0.f, 0.f, 0.f, 0.f, 0.f, 0.f};

    // xz for the *current* step, register-preloaded one step ahead
    float xzreg[8];
    int gph = (wv < 4) ? wv : 0;         // gate handled in phase 2
    int gcol = gph * HDIM + col;
    if (wv < 4) {
#pragma unroll
        for (int i = 0; i < 8; ++i) {
            int b = m * 16 + half * 8 + i;
            xzreg[i] = xz[(size_t)b * G4H + gcol];          // t = 0
        }
    }

    for (int t = 0; t < T_STEPS; ++t) {
        // phase 1: A loads + 16 back-to-back WMMAs (B already in registers)
        v16bf afr[4];
#pragma unroll
        for (int ks = 0; ks < 4; ++ks)
            afr[ks] = load_a16x32(h + (size_t)(m * 16) * HDIM + kbase + ks * 32,
                                  HDIM, lane);
        v8f acc[4] = {};
#pragma unroll
        for (int ks = 0; ks < 4; ++ks)
#pragma unroll
            for (int g = 0; g < 4; ++g)
                acc[g] = WMMA_BF16(afr[ks], bfr[g][ks], acc[g]);

#pragma unroll
        for (int g = 0; g < 4; ++g)
#pragma unroll
            for (int i = 0; i < 8; ++i)
                red[wv][g][i][lane] = acc[g][i];
        __syncthreads();

        // phase 2: waves 0..3 reduce one gate each + add preloaded xz;
        // then issue xz loads for t+1 (complete under the barrier wait)
        if (wv < 4) {
#pragma unroll
            for (int i = 0; i < 8; ++i) {
                float z = xzreg[i];
#pragma unroll
                for (int w = 0; w < 8; ++w)
                    z += red[w][gph][i][lane];
                red[0][gph][i][lane] = z;
            }
            if (t + 1 < T_STEPS) {
#pragma unroll
                for (int i = 0; i < 8; ++i) {
                    int b = m * 16 + half * 8 + i;
                    xzreg[i] = xz[(size_t)((t + 1) * BDIM + b) * G4H + gcol];
                }
            }
        }
        __syncthreads();

        // phase 3: wave 0 fused cell update (hardware tanh; c in registers)
        if (wv == 0) {
#pragma unroll
            for (int i = 0; i < 8; ++i) {
                int b = m * 16 + half * 8 + i;
                float si = fast_sigmoid(red[0][0][i][lane]);
                float sf = fast_sigmoid(red[0][1][i][lane]);
                float tg = FAST_TANH(red[0][2][i][lane]);
                float so = fast_sigmoid(red[0][3][i][lane]);
                float cn = sf * creg[i] + si * tg;
                creg[i] = cn;
                h[(size_t)b * HDIM + col] = (__bf16)(so * FAST_TANH(cn));
            }
        }
        __syncthreads();

        // grid-wide release/acquire barrier on a monotonic counter
        if (wv == 0) {
            __threadfence();
            if (lane == 0)
                __hip_atomic_fetch_add(ctr, 1u, __ATOMIC_RELEASE,
                                       __HIP_MEMORY_SCOPE_AGENT);
        }
        unsigned target = (unsigned)NSCAN_BLOCKS * (unsigned)(t + 1);
        if (lane == 0) {
            while (__hip_atomic_load(ctr, __ATOMIC_ACQUIRE,
                                     __HIP_MEMORY_SCOPE_AGENT) < target)
                __builtin_amdgcn_s_sleep(1);
        }
        __threadfence();
    }
}

// ---------------------------------------------------------------------------
// logits = h_last @ Wd + bd. Ping-pong pipeline, one 16x16 tile per wave.
__global__ void k_logits(const __bf16* __restrict__ h, const __bf16* __restrict__ WdT,
                         const float* __restrict__ bd, float* __restrict__ out) {
    int lane = threadIdx.x & 31;
    int wv   = threadIdx.x >> 5;
    int m    = wv & 1;
    int no   = wv >> 1;

    const __bf16* Abase = h + (size_t)(m * 16) * HDIM;
    const __bf16* Bbase = WdT + (size_t)(no * 16) * HDIM;

    v8f acc = {};
    v16bf a0, a1, b0, b1;
    a0 = load_a16x32(Abase, HDIM, lane);
    b0 = load_b32x16T(Bbase, HDIM, lane);
    for (int k0 = 0; k0 < HDIM - 64; k0 += 64) {
        a1 = load_a16x32(Abase + k0 + 32, HDIM, lane);
        b1 = load_b32x16T(Bbase + k0 + 32, HDIM, lane);
        acc = WMMA_BF16(a0, b0, acc);
        a0 = load_a16x32(Abase + k0 + 64, HDIM, lane);
        b0 = load_b32x16T(Bbase + k0 + 64, HDIM, lane);
        acc = WMMA_BF16(a1, b1, acc);
    }
    a1 = load_a16x32(Abase + HDIM - 32, HDIM, lane);
    b1 = load_b32x16T(Bbase + HDIM - 32, HDIM, lane);
    acc = WMMA_BF16(a0, b0, acc);
    acc = WMMA_BF16(a1, b1, acc);

    int nl = lane & 15, half = lane >> 4;
    int col = no * 16 + nl;
    float bias = bd[col];
#pragma unroll
    for (int i = 0; i < 8; ++i) {
        int b = m * 16 + half * 8 + i;
        out[(size_t)b * ODIM + col] = acc[i] + bias;
    }
}

// ---------------------------------------------------------------------------
extern "C" void kernel_launch(void* const* d_in, const int* in_sizes, int n_in,
                              void* d_out, int out_size, void* d_ws, size_t ws_size,
                              hipStream_t stream) {
    (void)in_sizes; (void)n_in; (void)out_size; (void)ws_size;

    const float* x  = (const float*)d_in[0];
    const float* Wi = (const float*)d_in[1];
    const float* Wh = (const float*)d_in[2];
    const float* bh = (const float*)d_in[3];
    const float* Wd = (const float*)d_in[4];
    const float* bd = (const float*)d_in[5];
    float* out = (float*)d_out;

    char* ws = (char*)d_ws;
    __bf16*   xb   = (__bf16*)(ws + OFF_XB);
    __bf16*   WiT  = (__bf16*)(ws + OFF_WIT);
    __bf16*   WhT  = (__bf16*)(ws + OFF_WHT);
    __bf16*   WdT  = (__bf16*)(ws + OFF_WDT);
    __bf16*   hbuf = (__bf16*)(ws + OFF_H);
    unsigned* ctr  = (unsigned*)(ws + OFF_CTR);
    float*    xz   = (float*)(ws + OFF_XZ);

    k_init_state<<<128, 256, 0, stream>>>(hbuf, ctr);
    k_cvt_x<<<(BDIM * T_STEPS * DDIM) / 256, 256, 0, stream>>>(x, xb, BDIM * T_STEPS * DDIM);
    k_transpose_cvt<<<(DDIM * G4H) / 256, 256, 0, stream>>>(Wi, WiT, DDIM, G4H);
    k_transpose_cvt<<<(HDIM * G4H) / 256, 256, 0, stream>>>(Wh, WhT, HDIM, G4H);
    k_transpose_cvt<<<(HDIM * ODIM) / 256, 256, 0, stream>>>(Wd, WdT, HDIM, ODIM);

    k_gemm_xz<<<8192, 256, 0, stream>>>(xb, WiT, bh, xz);
    k_lstm_scan<<<NSCAN_BLOCKS, 256, 0, stream>>>(xz, WhT, hbuf, ctr);
    k_logits<<<1, 512, 0, stream>>>(hbuf, WdT, bd, out);
}
